// BorzoiTrunk_29076928594412
// MI455X (gfx1250) — matrix operand order repaired and checked by reading, not compiled, not run on hardware
//
#include <hip/hip_runtime.h>

// ---------------------------------------------------------------------------
// Types for CDNA5 WMMA (wave32, 16x16x32 bf16 -> f32) and TDM descriptors
// ---------------------------------------------------------------------------
typedef __attribute__((ext_vector_type(16))) __bf16 v16bf;
typedef __attribute__((ext_vector_type(8)))  float  v8f;
typedef __attribute__((ext_vector_type(4)))  unsigned int u32x4;
typedef __attribute__((ext_vector_type(8)))  int i32x8;
typedef __attribute__((ext_vector_type(4)))  int i32x4;

#define DEV_INLINE __device__ __forceinline__

#if __has_builtin(__builtin_amdgcn_tensor_load_to_lds) && \
    __has_builtin(__builtin_amdgcn_s_wait_tensorcnt)
#define USE_TDM 1
#else
#define USE_TDM 0
#endif

DEV_INLINE float gelu_tanh(float x) {
  return 0.5f * x * (1.0f + tanhf(0.7978845608028654f * (x + 0.044715f * x * x * x)));
}

#if USE_TDM
// Issue a 2D TDM tile load: global (row-major, elem=2B) -> LDS, row-packed.
// OOB rows/cols (beyond tensor_rows/tensor_cols) are zero-filled by TDM.
DEV_INLINE void tdm_load_tile_2d(unsigned lds_off_bytes, const void* gptr,
                                 unsigned tile_cols, unsigned tile_rows,
                                 unsigned tensor_cols, unsigned tensor_rows,
                                 unsigned row_stride_elems)
{
  unsigned long long ga = (unsigned long long)gptr;
  u32x4 g0;
  g0[0] = 1u;                                            // count=1 (valid), user mode
  g0[1] = lds_off_bytes;                                 // lds_addr
  g0[2] = (unsigned)(ga & 0xffffffffu);                  // global_addr[31:0]
  g0[3] = (unsigned)((ga >> 32) & 0x01ffffffu) | (2u << 30);  // ga[56:32] | type=2
  i32x8 g1;
  g1[0] = (int)(1u << 16);                               // data_size=1 (2 bytes)
  g1[1] = (int)((tensor_cols & 0xffffu) << 16);          // tensor_dim0[15:0]
  g1[2] = (int)((tensor_cols >> 16) | ((tensor_rows & 0xffffu) << 16));
  g1[3] = (int)((tensor_rows >> 16) | ((tile_cols & 0xffffu) << 16));
  g1[4] = (int)(tile_rows & 0xffffu);                    // tile_dim1 (tile_dim2=0)
  g1[5] = (int)row_stride_elems;                         // tensor_dim0_stride[31:0]
  g1[6] = 0;
  g1[7] = 0;
  i32x4 z4; z4[0] = 0; z4[1] = 0; z4[2] = 0; z4[3] = 0;  // groups 2/3 unused (2D)
  i32x8 z8;
  #pragma unroll
  for (int i = 0; i < 8; ++i) z8[i] = 0;
  // 6-arg form (clang-23 / therock-10.0 signature)
  __builtin_amdgcn_tensor_load_to_lds(g0, g1, z4, z4, z8, 0);
}
#endif

// ---------------------------------------------------------------------------
// Generic bf16 WMMA GEMM:  C[M,N](f32) (+bias,act,residual) = A[M,K] * B[K,N]
//   - A row-major bf16 (lda), per-batch stride sA   (TDM-staged, double buffer)
//   - B row-major bf16 (ldb), per-batch stride sB   (manual transposed stage)
//     conv_cin > 0: logical row k -> (t = k/conv_cin, ci = k%conv_cin),
//                   element B[ci*ldb + n + t]  (implicit conv-as-GEMM, pad=2)
//   - C f32 (ldc, sC) optional; Dbf bf16 (same ldc,sC) optional
//   - bias: per-n (bias_per_m==0) or per-m (==1); act: 0 none, 1 relu, 2 gelu
//   - Cadd: residual added after bias/act (same ldc/sC as C)
// Workgroup: 256 threads = 8 waves; tile 256(M) x 64(N) x 32(K).
// Each wave: rows [32w,32w+32): 2 A-frags x 4 B-frags = 8 wmma per K-step.
// Software pipeline: tile s+1 is TDM'd / staged while tile s is computed.
// ---------------------------------------------------------------------------
__global__ __launch_bounds__(256) void k_gemm_bf16(
    const __bf16* __restrict__ A, long long lda, long long sA,
    const __bf16* __restrict__ B, long long ldb, long long sB,
    float* __restrict__ C, long long ldc, long long sC,
    int M, int N, int K,
    const float* __restrict__ bias, int bias_per_m,
    const float* __restrict__ Cadd,
    __bf16* __restrict__ Dbf,
    int conv_cin, int act)
{
  __shared__ __align__(16) __bf16 sAt[2][256][32];   // [buf][m][k] (TDM dest)
  __shared__ __align__(16) __bf16 sBt[2][64][32];    // [buf][n][k] (transposed)

  const int tid  = threadIdx.x;
  const int wave = tid >> 5;
  const int lane = tid & 31;
  const int hi   = lane >> 4;   // lane half: selects K sub-slab
  const int lo   = lane & 15;

  const int m0 = blockIdx.y * 256;
  const int n0 = blockIdx.x * 64;
  const long long zb = blockIdx.z;

  const __bf16* Ab = A + zb * sA;
  const __bf16* Bb = B + zb * sB;

#if !USE_TDM
  // fallback: manual A staging, zero-filled M edge
  auto stageA = [&](int k0, int buf) {
    #pragma unroll
    for (int it = 0; it < 4; ++it) {
      int task = tid + it * 256;
      int row  = task >> 2;
      int seg  = task & 3;
      uint4 val = make_uint4(0u, 0u, 0u, 0u);
      if (m0 + row < M)
        val = *(const uint4*)(Ab + (long long)(m0 + row) * lda + (k0 + seg * 8));
      *(uint4*)&sAt[buf][row][seg * 8] = val;
    }
  };
#endif

  // B tile: transpose-stage [32 k][64 n] global -> [64 n][32 k] LDS
  auto stageB = [&](int k0, int buf) {
    int k   = tid >> 3;
    int seg = tid & 7;
    int kk  = k0 + k;
    const __bf16* src;
    if (conv_cin > 0) {
      int t  = kk / conv_cin;           // tap index
      int ci = kk - t * conv_cin;
      src = Bb + (long long)ci * ldb + (n0 + seg * 8 + t);
    } else {
      src = Bb + (long long)kk * ldb + (n0 + seg * 8);
    }
    #pragma unroll
    for (int e = 0; e < 8; ++e) {
      int n = seg * 8 + e;
      __bf16 vv = (__bf16)0.0f;
      if (n0 + n < N) vv = src[e];
      sBt[buf][n][k] = vv;
    }
  };

  union CF { v8f v; float e[8]; };
  CF acc[2][4];
  #pragma unroll
  for (int s = 0; s < 2; ++s)
    #pragma unroll
    for (int c = 0; c < 4; ++c)
      #pragma unroll
      for (int r = 0; r < 8; ++r) acc[s][c].e[r] = 0.0f;

  const int steps = K >> 5;   // K is always a multiple of 32 here

  // ---- prologue: bring in tile 0 ----
#if USE_TDM
  if (wave == 0) {
    tdm_load_tile_2d((unsigned)(unsigned long long)(void*)&sAt[0][0][0],
                     Ab + (long long)m0 * lda,
                     /*tile_cols=*/32, /*tile_rows=*/256,
                     /*tensor_cols=*/(unsigned)K,
                     /*tensor_rows=*/(unsigned)(M - m0),
                     /*row_stride_elems=*/(unsigned)lda);
  }
#else
  stageA(0, 0);
#endif
  stageB(0, 0);
#if USE_TDM
  if (wave == 0) __builtin_amdgcn_s_wait_tensorcnt(0);
#endif
  __syncthreads();

  for (int s = 0; s < steps; ++s) {
    const int cur = s & 1;
    const int nxt = cur ^ 1;
    const int k0  = s << 5;

    // ---- prefetch tile s+1 into the alternate buffers (overlaps compute) ----
    if (s + 1 < steps) {
#if USE_TDM
      if (wave == 0) {
        tdm_load_tile_2d((unsigned)(unsigned long long)(void*)&sAt[nxt][0][0],
                         Ab + (long long)m0 * lda + (k0 + 32),
                         /*tile_cols=*/32, /*tile_rows=*/256,
                         /*tensor_cols=*/(unsigned)(K - (k0 + 32)),
                         /*tensor_rows=*/(unsigned)(M - m0),
                         /*row_stride_elems=*/(unsigned)lda);
      }
#else
      stageA(k0 + 32, nxt);
#endif
      stageB(k0 + 32, nxt);
    }

    // ---- fragments per ISA 7.12.2 lane layouts, compute on tile s ----
    union F16 { v16bf v; uint4 q[2]; };
    F16 af[2];
    #pragma unroll
    for (int ss = 0; ss < 2; ++ss) {
      const int mrow = wave * 32 + ss * 16 + lo;
      // A: lanes 0-15 hold K {0..7,16..23}; lanes 16-31 hold K {8..15,24..31}
      af[ss].q[0] = *(const uint4*)&sAt[cur][mrow][hi * 8];
      af[ss].q[1] = *(const uint4*)&sAt[cur][mrow][16 + hi * 8];
    }
    #pragma unroll
    for (int c = 0; c < 4; ++c) {
      F16 bfrag;
      const int ncol = c * 16 + lo;
      // B: lanes 0-15 hold K {0..7,16..23}; lanes 16-31 hold K {8..15,24..31}
      bfrag.q[0] = *(const uint4*)&sBt[cur][ncol][hi * 8];
      bfrag.q[1] = *(const uint4*)&sBt[cur][ncol][16 + hi * 8];
      acc[0][c].v = __builtin_amdgcn_wmma_f32_16x16x32_bf16(
          false, af[0].v, false, bfrag.v, (short)0, acc[0][c].v, false, false);
      acc[1][c].v = __builtin_amdgcn_wmma_f32_16x16x32_bf16(
          false, af[1].v, false, bfrag.v, (short)0, acc[1][c].v, false, false);
    }

    // ---- flip buffers: drain next tile's TDM, fence staging stores ----
    if (s + 1 < steps) {
#if USE_TDM
      if (wave == 0) __builtin_amdgcn_s_wait_tensorcnt(0);
#endif
      __syncthreads();
    }
  }

  // ---- epilogue: C layout lanes0-15 M=r, lanes16-31 M=8+r; N=lane&15 ----
  #pragma unroll
  for (int s = 0; s < 2; ++s) {
    #pragma unroll
    for (int c = 0; c < 4; ++c) {
      #pragma unroll
      for (int r = 0; r < 8; ++r) {
        int m = m0 + wave * 32 + s * 16 + hi * 8 + r;
        int n = n0 + c * 16 + lo;
        if (m < M && n < N) {
          float v = acc[s][c].e[r];
          if (bias) v += bias[bias_per_m ? m : n];
          if (act == 1)      v = fmaxf(v, 0.0f);
          else if (act == 2) v = gelu_tanh(v);
          long long o = zb * sC + (long long)m * ldc + n;
          if (Cadd) v += Cadd[o];
          if (C)   C[o]   = v;
          if (Dbf) Dbf[o] = (__bf16)v;
        }
      }
    }
  }
}

// ---------------------------------------------------------------------------
// Stem: conv1d(4->512, k=15, same) fused with maxpool2.  out [512][Lout] f32.
// ---------------------------------------------------------------------------
__global__ void k_stem(const float* __restrict__ x, const float* __restrict__ w,
                       const float* __restrict__ b, float* __restrict__ out,
                       int Lin, int Lout)
{
  long long idx = (long long)blockIdx.x * 256 + threadIdx.x;
  if (idx >= (long long)512 * Lout) return;
  int c = (int)(idx / Lout);
  int l = (int)(idx % Lout);
  float best = -3.4e38f;
  #pragma unroll
  for (int s = 0; s < 2; ++s) {
    int l2 = 2 * l + s;
    float accv = b[c];
    for (int ci = 0; ci < 4; ++ci) {
      const float* wr = w + (c * 4 + ci) * 15;
      const float* xr = x + (long long)ci * Lin;
      #pragma unroll
      for (int t = 0; t < 15; ++t) {
        int p = l2 + t - 7;
        if (p >= 0 && p < Lin) accv += wr[t] * xr[p];
      }
    }
    best = fmaxf(best, accv);
  }
  out[idx] = best;
}

// maxpool2: in [C][2*Lout] -> out [C][Lout]
__global__ void k_maxpool2(const float* __restrict__ in, float* __restrict__ out,
                           int C, int Lout)
{
  long long idx = (long long)blockIdx.x * 256 + threadIdx.x;
  if (idx >= (long long)C * Lout) return;
  int c = (int)(idx / Lout);
  int l = (int)(idx % Lout);
  const float* r = in + (long long)c * (2 * Lout);
  out[idx] = fmaxf(r[2 * l], r[2 * l + 1]);
}

// BN(eval) -> GELU(tanh) -> bf16, written into a 2-column zero-padded buffer
// out [C][L+4] bf16, cols [2, L+2) hold data.
__global__ void k_bngelu_pad(const float* __restrict__ x,
                             const float* __restrict__ g, const float* __restrict__ b,
                             const float* __restrict__ m, const float* __restrict__ v,
                             __bf16* __restrict__ out, int C, int L)
{
  long long idx = (long long)blockIdx.x * 256 + threadIdx.x;
  int Lp = L + 4;
  if (idx >= (long long)C * Lp) return;
  int ci = (int)(idx / Lp);
  int p  = (int)(idx % Lp);
  __bf16 r = (__bf16)0.0f;
  if (p >= 2 && p < L + 2) {
    float xv = x[(long long)ci * L + (p - 2)];
    float sc = g[ci] * rsqrtf(v[ci] + 1e-3f);
    float xh = (xv - m[ci]) * sc + b[ci];
    r = (__bf16)gelu_tanh(xh);
  }
  out[idx] = r;
}

// W [Cout][Cin][5] f32 -> bf16 [Cout][5*Cin] with k = t*Cin + ci
__global__ void k_convw_bf16(const float* __restrict__ w, __bf16* __restrict__ out,
                             int Co, int Ci)
{
  long long idx = (long long)blockIdx.x * 256 + threadIdx.x;
  long long total = (long long)Co * Ci * 5;
  if (idx >= total) return;
  int K5 = 5 * Ci;
  int co = (int)(idx / K5);
  int k  = (int)(idx % K5);
  int t  = k / Ci;
  int ci = k - t * Ci;
  out[idx] = (__bf16)w[((long long)co * Ci + ci) * 5 + t];
}

__global__ void k_f32_to_bf16(const float* __restrict__ in, __bf16* __restrict__ out,
                              long long n)
{
  long long idx = (long long)blockIdx.x * 256 + threadIdx.x;
  if (idx < n) out[idx] = (__bf16)in[idx];
}

// transpose f32 [R][C] -> bf16 [C][R]
__global__ void k_transpose_f32_to_bf16(const float* __restrict__ in,
                                        __bf16* __restrict__ out, int R, int C)
{
  long long idx = (long long)blockIdx.x * 256 + threadIdx.x;
  if (idx >= (long long)R * C) return;
  int r = (int)(idx / C);
  int c = (int)(idx % C);
  out[(long long)c * R + r] = (__bf16)in[idx];
}

// transpose f32 [R][C] -> f32 [C][R]
__global__ void k_transpose_f32(const float* __restrict__ in, float* __restrict__ out,
                                int R, int C)
{
  long long idx = (long long)blockIdx.x * 256 + threadIdx.x;
  if (idx >= (long long)R * C) return;
  int r = (int)(idx / C);
  int c = (int)(idx % C);
  out[(long long)c * R + r] = in[idx];
}

// LayerNorm over D=1536 per row, output bf16. One block (256 thr) per row.
__global__ void k_ln_bf16(const float* __restrict__ x, const float* __restrict__ g,
                          const float* __restrict__ b, __bf16* __restrict__ out, int D)
{
  __shared__ float red[256];
  __shared__ float sh_mu, sh_rstd;
  int row = blockIdx.x, tid = threadIdx.x;
  const float* xr = x + (long long)row * D;
  float s = 0.0f;
  for (int c = tid; c < D; c += 256) s += xr[c];
  red[tid] = s; __syncthreads();
  for (int o = 128; o > 0; o >>= 1) { if (tid < o) red[tid] += red[tid + o]; __syncthreads(); }
  if (tid == 0) sh_mu = red[0] / (float)D;
  __syncthreads();
  float mu = sh_mu, v = 0.0f;
  for (int c = tid; c < D; c += 256) { float d = xr[c] - mu; v += d * d; }
  red[tid] = v; __syncthreads();
  for (int o = 128; o > 0; o >>= 1) { if (tid < o) red[tid] += red[tid + o]; __syncthreads(); }
  if (tid == 0) sh_rstd = rsqrtf(red[0] / (float)D + 1e-3f);
  __syncthreads();
  float r = sh_rstd;
  for (int c = tid; c < D; c += 256)
    out[(long long)row * D + c] = (__bf16)((xr[c] - mu) * r * g[c] + b[c]);
}

// central-mask positional features, [2N-1][32]
__global__ void k_posembed(float* __restrict__ pos, int N)
{
  long long idx = (long long)blockIdx.x * 256 + threadIdx.x;
  long long total = (long long)(2 * N - 1) * 32;
  if (idx >= total) return;
  int i = (int)(idx / 32);
  int f = (int)(idx % 32);
  float dist = (float)(i - (N - 1));
  float pr = expf(logf((float)N + 1.0f) / 16.0f);
  int j = f & 15;
  float cw = powf(pr, (float)(j + 1)) - 1.0f;
  float e = (cw > fabsf(dist)) ? 1.0f : 0.0f;
  if (f >= 16) {
    float sg = dist > 0.0f ? 1.0f : (dist < 0.0f ? -1.0f : 0.0f);
    e *= sg;
  }
  pos[idx] = e;
}

// q -> (q*scale + rcb) and (q*scale + rpb), bf16.  q [2048][512].
__global__ void k_qprep(const float* __restrict__ q, const float* __restrict__ rcb,
                        const float* __restrict__ rpb, __bf16* __restrict__ qc,
                        __bf16* __restrict__ qr, long long total)
{
  long long idx = (long long)blockIdx.x * 256 + threadIdx.x;
  if (idx >= total) return;
  int hd = (int)(idx % 512);
  float qs = q[idx] * 0.125f;   // DK^-0.5, DK=64
  qc[idx] = (__bf16)(qs + rcb[hd]);
  qr[idx] = (__bf16)(qs + rpb[hd]);
}

// softmax over j of content[h][i][j] + rel_full[h][i][N-1+j-i]; out bf16 probs
__global__ void k_attn_softmax(const float* __restrict__ cont,
                               const float* __restrict__ relf,
                               __bf16* __restrict__ attnp, int N)
{
  __shared__ float red[256];
  int i = blockIdx.x, h = blockIdx.y, tid = threadIdx.x;
  const float* cr = cont + ((long long)h * N + i) * N;
  const float* rr = relf + ((long long)h * N + i) * (2LL * N - 1);
  __bf16* orow = attnp + ((long long)h * N + i) * N;
  float mx = -3.4e38f;
  for (int j = tid; j < N; j += 256)
    mx = fmaxf(mx, cr[j] + rr[N - 1 + j - i]);
  red[tid] = mx; __syncthreads();
  for (int o = 128; o > 0; o >>= 1) { if (tid < o) red[tid] = fmaxf(red[tid], red[tid + o]); __syncthreads(); }
  mx = red[0]; __syncthreads();
  float sum = 0.0f;
  for (int j = tid; j < N; j += 256)
    sum += expf(cr[j] + rr[N - 1 + j - i] - mx);
  red[tid] = sum; __syncthreads();
  for (int o = 128; o > 0; o >>= 1) { if (tid < o) red[tid] += red[tid + o]; __syncthreads(); }
  float inv = 1.0f / red[0];
  for (int j = tid; j < N; j += 256)
    orow[j] = (__bf16)(expf(cr[j] + rr[N - 1 + j - i] - mx) * inv);
}

// ---------------------------------------------------------------------------
// Host orchestration
// ---------------------------------------------------------------------------
extern "C" void kernel_launch(void* const* d_in, const int* in_sizes, int n_in,
                              void* d_out, int out_size, void* d_ws, size_t ws_size,
                              hipStream_t stream)
{
  (void)in_sizes; (void)n_in; (void)out_size; (void)ws_size;

  // ---- input map (setup_inputs dict flattening) ----
  const float* x      = (const float*)d_in[0];
  const float* stem_w = (const float*)d_in[1];
  const float* stem_b = (const float*)d_in[2];
  // tower i: 3+6i .. 8+6i (bn_g, bn_b, bn_m, bn_v, w, b); unet1 at 33..38
  const float* ln1_g = (const float*)d_in[39];
  const float* ln1_b = (const float*)d_in[40];
  const float* wq    = (const float*)d_in[41];
  const float* wk    = (const float*)d_in[42];
  const float* wv    = (const float*)d_in[43];
  const float* wo    = (const float*)d_in[44];
  const float* wo_b  = (const float*)d_in[45];
  const float* wrk   = (const float*)d_in[46];
  const float* rcb   = (const float*)d_in[47];
  const float* rpb   = (const float*)d_in[48];
  const float* ln2_g = (const float*)d_in[49];
  const float* ln2_b = (const float*)d_in[50];
  const float* ff1_w = (const float*)d_in[51];
  const float* ff1_b = (const float*)d_in[52];
  const float* ff2_w = (const float*)d_in[53];
  const float* ff2_b = (const float*)d_in[54];

  // ---- workspace bump allocator ----
  size_t off = 0;
  auto balloc = [&](size_t bytes) -> char* {
    char* p = (char*)d_ws + off;
    off += (bytes + 255) & ~(size_t)255;
    return p;
  };
  float*  A0     = (float*) balloc(40ull * 1024 * 1024 * 4);   // activation ping
  float*  A1     = (float*) balloc(40ull * 1024 * 1024 * 4);   // activation pong
  __bf16* XH     = (__bf16*)balloc(34ull * 1024 * 1024 * 2);   // padded bf16 acts
  __bf16* WBF    = (__bf16*)balloc(10ull * 1024 * 1024 * 2);   // bf16 weights
  __bf16* XLN    = (__bf16*)balloc(2048ull * 1536 * 2);
  float*  Qf     = (float*) balloc(2048ull * 512 * 4);
  float*  Kf     = (float*) balloc(2048ull * 512 * 4);
  float*  Vf     = (float*) balloc(2048ull * 1536 * 4);
  __bf16* QC     = (__bf16*)balloc(2048ull * 512 * 2);
  __bf16* QR     = (__bf16*)balloc(2048ull * 512 * 2);
  __bf16* KT     = (__bf16*)balloc(512ull * 2048 * 2);
  __bf16* VBF    = (__bf16*)balloc(2048ull * 1536 * 2);
  float*  POSF   = (float*) balloc(4095ull * 32 * 4);
  __bf16* POSBF  = (__bf16*)balloc(4095ull * 32 * 2);
  float*  RELK   = (float*) balloc(4095ull * 512 * 4);
  __bf16* RELKT  = (__bf16*)balloc(512ull * 4095 * 2);
  float*  CONT   = (float*) balloc(8ull * 2048 * 2048 * 4);
  float*  RELF   = (float*) balloc(8ull * 2048 * 4095 * 4);
  __bf16* ATTP   = (__bf16*)balloc(8ull * 2048 * 2048 * 2);
  __bf16* ATTOBF = (__bf16*)balloc(2048ull * 1536 * 2);
  __bf16* FFMID  = (__bf16*)balloc(2048ull * 3072 * 2);

  auto blocks = [](long long n) { return (unsigned)((n + 255) / 256); };

  auto gemm = [&](const __bf16* A, long long lda, long long sA,
                  const __bf16* B, long long ldb, long long sB,
                  float* C, long long ldc, long long sC,
                  int M, int N, int K,
                  const float* bias, int bias_per_m,
                  const float* Cadd, __bf16* Dbf,
                  int conv_cin, int act, int batch) {
    dim3 g((unsigned)((N + 63) / 64), (unsigned)((M + 255) / 256), (unsigned)batch);
    k_gemm_bf16<<<g, 256, 0, stream>>>(A, lda, sA, B, ldb, sB, C, ldc, sC,
                                       M, N, K, bias, bias_per_m, Cadd, Dbf,
                                       conv_cin, act);
  };

  auto cvt = [&](const float* src, __bf16* dst, long long n) {
    k_f32_to_bf16<<<blocks(n), 256, 0, stream>>>(src, dst, n);
  };

  auto conv_block = [&](const float* in, float* out, int Ci, int Co, int L,
                        const float* bng, const float* bnb, const float* bnm,
                        const float* bnv, const float* W, const float* bb) {
    long long totBG = (long long)Ci * (L + 4);
    k_bngelu_pad<<<blocks(totBG), 256, 0, stream>>>(in, bng, bnb, bnm, bnv, XH, Ci, L);
    long long totW = (long long)Co * Ci * 5;
    k_convw_bf16<<<blocks(totW), 256, 0, stream>>>(W, WBF, Co, Ci);
    gemm(WBF, 5LL * Ci, 0, XH, (long long)L + 4, 0, out, L, 0,
         Co, L, 5 * Ci, bb, /*bias_per_m=*/1, nullptr, nullptr, /*conv_cin=*/Ci, 0, 1);
  };

  // ================= conv trunk =================
  int Ls = 65536;
  k_stem<<<blocks(512LL * Ls), 256, 0, stream>>>(x, stem_w, stem_b, A0, 131072, Ls);

  float* cur = A0;
  float* nxt = A1;
  const int ci_arr[5] = {512, 608, 736, 896, 1056};
  const int co_arr[5] = {608, 736, 896, 1056, 1280};
  for (int i = 0; i < 5; ++i) {
    conv_block(cur, nxt, ci_arr[i], co_arr[i], Ls,
               (const float*)d_in[3 + 6 * i], (const float*)d_in[4 + 6 * i],
               (const float*)d_in[5 + 6 * i], (const float*)d_in[6 + 6 * i],
               (const float*)d_in[7 + 6 * i], (const float*)d_in[8 + 6 * i]);
    if (i < 4) {
      k_maxpool2<<<blocks((long long)co_arr[i] * (Ls / 2)), 256, 0, stream>>>(
          nxt, cur, co_arr[i], Ls / 2);
      Ls /= 2;                       // cur holds pooled [Co][Ls]
    } else {
      float* t = cur; cur = nxt; nxt = t;   // cur holds [1280][4096]
    }
  }
  // unet1 pool then conv block -> [1536][2048]
  k_maxpool2<<<blocks(1280LL * (Ls / 2)), 256, 0, stream>>>(cur, nxt, 1280, Ls / 2);
  Ls /= 2;
  { float* t = cur; cur = nxt; nxt = t; }
  conv_block(cur, nxt, 1280, 1536, Ls,
             (const float*)d_in[33], (const float*)d_in[34], (const float*)d_in[35],
             (const float*)d_in[36], (const float*)d_in[37], (const float*)d_in[38]);
  // transpose [1536][2048] -> x state in d_out [2048][1536]
  float* X = (float*)d_out;
  k_transpose_f32<<<blocks(1536LL * 2048), 256, 0, stream>>>(nxt, X, 1536, 2048);

  // ================= transformer =================
  const int N = 2048, D = 1536;
  k_posembed<<<blocks(4095LL * 32), 256, 0, stream>>>(POSF, N);
  cvt(POSF, POSBF, 4095LL * 32);

  for (int d = 0; d < 4; ++d) {
    // --- attention ---
    k_ln_bf16<<<N, 256, 0, stream>>>(X, ln1_g + d * D, ln1_b + d * D, XLN, D);

    cvt(wq + (long long)d * D * 512, WBF, (long long)D * 512);
    gemm(XLN, D, 0, WBF, 512, 0, Qf, 512, 0, N, 512, D, nullptr, 0, nullptr, nullptr, 0, 0, 1);
    cvt(wk + (long long)d * D * 512, WBF, (long long)D * 512);
    gemm(XLN, D, 0, WBF, 512, 0, Kf, 512, 0, N, 512, D, nullptr, 0, nullptr, nullptr, 0, 0, 1);
    cvt(wv + (long long)d * D * D, WBF, (long long)D * D);
    gemm(XLN, D, 0, WBF, D, 0, Vf, D, 0, N, D, D, nullptr, 0, nullptr, nullptr, 0, 0, 1);

    k_qprep<<<blocks((long long)N * 512), 256, 0, stream>>>(
        Qf, rcb + d * 512, rpb + d * 512, QC, QR, (long long)N * 512);
    k_transpose_f32_to_bf16<<<blocks((long long)N * 512), 256, 0, stream>>>(Kf, KT, N, 512);
    cvt(Vf, VBF, (long long)N * D);

    // content[h] = (q*s + rcb) @ k^T    (8 heads, K=64)
    gemm(QC, 512, 64, KT, N, 64LL * N, CONT, N, (long long)N * N,
         N, N, 64, nullptr, 0, nullptr, nullptr, 0, 0, 8);

    // rel_k = positions @ wrk ; rel_full[h] = (q*s + rpb) @ rel_k^T
    cvt(wrk + (long long)d * 32 * 512, WBF, 32LL * 512);
    gemm(POSBF, 32, 0, WBF, 512, 0, RELK, 512, 0, 4095, 512, 32,
         nullptr, 0, nullptr, nullptr, 0, 0, 1);
    k_transpose_f32_to_bf16<<<blocks(4095LL * 512), 256, 0, stream>>>(RELK, RELKT, 4095, 512);
    gemm(QR, 512, 64, RELKT, 4095, 64LL * 4095, RELF, 4095, (long long)N * 4095,
         N, 4095, 64, nullptr, 0, nullptr, nullptr, 0, 0, 8);

    // softmax with fused relative shift -> bf16 probs
    k_attn_softmax<<<dim3(N, 8), 256, 0, stream>>>(CONT, RELF, ATTP, N);

    // attn @ v (per-head strided view of v), emit bf16 directly
    gemm(ATTP, N, (long long)N * N, VBF, D, 192, nullptr, D, 192,
         N, 192, N, nullptr, 0, nullptr, ATTOBF, 0, 0, 8);

    // out proj + bias + residual into X
    cvt(wo + (long long)d * D * D, WBF, (long long)D * D);
    gemm(ATTOBF, D, 0, WBF, D, 0, X, D, 0, N, D, D,
         wo_b + d * D, 0, X, nullptr, 0, 0, 1);

    // --- feed-forward ---
    k_ln_bf16<<<N, 256, 0, stream>>>(X, ln2_g + d * D, ln2_b + d * D, XLN, D);
    cvt(ff1_w + (long long)d * D * 3072, WBF, (long long)D * 3072);
    gemm(XLN, D, 0, WBF, 3072, 0, nullptr, 3072, 0, N, 3072, D,
         ff1_b + d * 3072, 0, nullptr, FFMID, 0, /*relu*/1, 1);
    cvt(ff2_w + (long long)d * 3072 * D, WBF, 3072LL * D);
    gemm(FFMID, 3072, 0, WBF, D, 0, X, D, 0, N, D, 3072,
         ff2_b + d * D, 0, X, nullptr, 0, 0, 1);
  }
}
